// CPLSTM_20959440405049
// MI455X (gfx1250) — matrix operand order repaired and checked by reading, not compiled, not run on hardware
//
#include <hip/hip_runtime.h>
#include <cstdint>
#include <cstddef>

// Problem dims (from reference): S=128, B=32, D=1024, H=1024, 4R=1024, V=32000
#define S_LEN 128
#define B_SZ  32
#define D_DIM 1024
#define H_DIM 1024
#define R4    1024
#define RNK   256
#define V_SZ  32000

typedef __attribute__((ext_vector_type(16))) __bf16 v16bf;
typedef __attribute__((ext_vector_type(8)))  float  v8f;
typedef __attribute__((ext_vector_type(4)))  unsigned int v4u;
typedef __attribute__((ext_vector_type(8)))  int    v8i;
typedef __attribute__((ext_vector_type(4)))  int    v4i;

// ---------------------------------------------------------------------------
// WMMA helpers (CDNA5 16x16x32 bf16, f32 accumulate)
// ---------------------------------------------------------------------------
__device__ inline v8f wmma_bf16(v16bf a, v16bf b, v8f c) {
  return __builtin_amdgcn_wmma_f32_16x16x32_bf16(false, a, false, b, (short)0, c,
                                                 false, false);
}

// A-fragment: 16(M) x 32(K) bf16, row-major source with leading dim ld.
__device__ inline v16bf frag_a(const __bf16* base, int ld) {
  const int lane = threadIdx.x & 31;
  const __bf16* p = base + (lane & 15) * ld;
  const int kh = (lane >> 4) ? 16 : 0;
  v16bf f;
#pragma unroll
  for (int j = 0; j < 8; ++j) {
    int k = kh + (j << 1);
    f[2 * j]     = p[k];
    f[2 * j + 1] = p[k + 1];
  }
  return f;
}

// B-fragment: 32(K) x 16(N) bf16 from [N][K] storage, 16 contiguous per lane.
__device__ inline v16bf frag_b(const __bf16* base, int ld) {
  const int lane = threadIdx.x & 31;
  const __bf16* p = base + (lane & 15) * ld + ((lane >> 4) << 4);
  v16bf f;
#pragma unroll
  for (int j = 0; j < 16; ++j) f[j] = p[j];
  return f;
}

__device__ inline v8f vzero8() {
  v8f z;
#pragma unroll
  for (int i = 0; i < 8; ++i) z[i] = 0.0f;
  return z;
}

__device__ inline float sigf(float x) { return 1.0f / (1.0f + __expf(-x)); }

// Low 32 bits of a generic shared-memory pointer = wave-relative LDS byte
// address (flat->LDS truncation, ISA 10.2).
__device__ inline uint32_t lds_addr32(const void* p) {
  return (uint32_t)(uintptr_t)p;
}

// Async 128-bit global -> LDS copy (ASYNCcnt-tracked, no VGPR round trip).
__device__ inline void async_copy_b128(uint32_t lds_dst, const void* gsrc) {
  asm volatile("global_load_async_to_lds_b128 %0, %1, off"
               :: "v"(lds_dst), "v"(gsrc)
               : "memory");
}
__device__ inline void wait_async0() {
  asm volatile("s_wait_asynccnt 0x0" ::: "memory");
}

// TDM: DMA a 64-row x 1024-col bf16 tile (contiguous rows, stride 1024) from
// global into LDS. D# layout per CDNA5 ISA section 8 (2D tensor, groups 2/3
// zeroed / disabled). 6-arg builtin form (clang-23 / therock headers).
__device__ inline void tdm_load_64x1024_bf16(uint32_t lds_dst,
                                             unsigned long long gaddr) {
  v4u g0;
  g0[0] = 1u;                                          // count=1, user mode
  g0[1] = lds_dst;                                     // lds_addr
  g0[2] = (unsigned int)gaddr;                         // global_addr[31:0]
  g0[3] = (unsigned int)((gaddr >> 32) & 0x1FFFFFFull) // global_addr[56:32]
          | (2u << 30);                                // type=2 ("image")
  v8i g1;
  g1[0] = (int)(1u << 16);      // data_size=1 -> 2 bytes
  g1[1] = (int)(1024u << 16);   // tensor_dim0[15:0]  (bits 79:48) = 1024
  g1[2] = (int)(64u << 16);     // tensor_dim1[15:0]  (bits 111:80) = 64
  g1[3] = (int)(1024u << 16);   // tile_dim0          (bits 127:112) = 1024
  g1[4] = (int)64;              // tile_dim1 = 64, tile_dim2 = 0
  g1[5] = (int)1024;            // tensor_dim0_stride[31:0] = 1024
  g1[6] = 0;                    // stride hi / dim1_stride lo
  g1[7] = 0;
  v4i z4 = {0, 0, 0, 0};
  v8i z8 = {0, 0, 0, 0, 0, 0, 0, 0};
  __builtin_amdgcn_tensor_load_to_lds(g0, g1, z4, z4, z8, 0);
}

// ---------------------------------------------------------------------------
// Prep: f32 MxN -> bf16 NxM transpose (weights to [N][K] layout)
// ---------------------------------------------------------------------------
__global__ void k_transpose_cvt(const float* __restrict__ in,
                                __bf16* __restrict__ out, int M, int N) {
  __shared__ float tile[32][33];
  const int bx = blockIdx.x * 32;  // N
  const int by = blockIdx.y * 32;  // M
  const int tx = threadIdx.x, ty = threadIdx.y;
#pragma unroll
  for (int i = ty; i < 32; i += 8)
    tile[i][tx] = in[(size_t)(by + i) * N + (bx + tx)];
  __syncthreads();
#pragma unroll
  for (int i = ty; i < 32; i += 8)
    out[(size_t)(bx + i) * M + (by + tx)] = (__bf16)tile[tx][i];
}

// ---------------------------------------------------------------------------
// Prep: gather embeddings -> bf16 x[sb][d]
// ---------------------------------------------------------------------------
__global__ void k_gather(const int* __restrict__ inp,
                         const float* __restrict__ emb,
                         __bf16* __restrict__ xbf) {
  const int sb = blockIdx.x;
  const int idx = inp[sb];
  const float4* src = (const float4*)(emb + (size_t)idx * D_DIM);
  __bf16* dst = xbf + (size_t)sb * D_DIM;
  const int t = threadIdx.x;  // 256
  float4 v = src[t];
  dst[4 * t + 0] = (__bf16)v.x;
  dst[4 * t + 1] = (__bf16)v.y;
  dst[4 * t + 2] = (__bf16)v.z;
  dst[4 * t + 3] = (__bf16)v.w;
}

// ---------------------------------------------------------------------------
// Prep: zero recurrent state + grid-sync counter
// ---------------------------------------------------------------------------
__global__ void k_init(__bf16* __restrict__ h, __bf16* __restrict__ g,
                       int* __restrict__ cnt) {
  const int t = blockIdx.x * blockDim.x + threadIdx.x;
  if (t < B_SZ * H_DIM) {
    h[t] = (__bf16)0.0f;
    g[t] = (__bf16)0.0f;
  }
  if (t == 0) *cnt = 0;
}

// ---------------------------------------------------------------------------
// xb = x @ b : (4096 x 1024) @ (1024 x 1024), bf16 WMMA, f32 out.
// Block tile 64M x 64N, 256 threads (8 waves), wave = 16M x 32N.
// A tile staged to LDS with async copies overlapped with B fragment loads.
// ---------------------------------------------------------------------------
__global__ void k_xb_gemm(const __bf16* __restrict__ xbf,
                          const __bf16* __restrict__ bT,  // [n][d]
                          float* __restrict__ xb) {
  __shared__ __bf16 as[64 * 32];  // A tile, ld 32
  const int tid = threadIdx.x;
  const int wave = tid >> 5;
  const int m_t = wave & 3;        // 0..3
  const int ngrp = wave >> 2;      // 0..1
  const int m0 = blockIdx.x * 64;
  const int n0 = blockIdx.y * 64;
  const int row = tid >> 2, seg = tid & 3;
  const uint32_t as_dst = lds_addr32(as + row * 32 + seg * 8);

  v8f acc[2];
  acc[0] = vzero8();
  acc[1] = vzero8();

  for (int kb = 0; kb < D_DIM / 32; ++kb) {
    const int k0 = kb * 32;
    // kick async A-tile fill (16B per thread)
    async_copy_b128(as_dst, xbf + (size_t)(m0 + row) * D_DIM + k0 + seg * 8);
    // overlap: fetch B fragments straight from L2 while LDS fill is in flight
    v16bf bfr0 = frag_b(bT + (size_t)(n0 + ngrp * 32) * D_DIM + k0, D_DIM);
    v16bf bfr1 = frag_b(bT + (size_t)(n0 + ngrp * 32 + 16) * D_DIM + k0, D_DIM);
    wait_async0();
    __syncthreads();
    v16bf af = frag_a(as + m_t * 16 * 32, 32);
    acc[0] = wmma_bf16(af, bfr0, acc[0]);
    acc[1] = wmma_bf16(af, bfr1, acc[1]);
    __syncthreads();
  }

  const int lane = tid & 31;
  const int col = lane & 15;
  const int rbase = (lane >> 4) * 8;
#pragma unroll
  for (int nt = 0; nt < 2; ++nt) {
    const int nc = n0 + ngrp * 32 + nt * 16 + col;
#pragma unroll
    for (int r = 0; r < 8; ++r) {
      const int rg = m0 + m_t * 16 + rbase + r;
      xb[(size_t)rg * R4 + nc] = acc[nt][r];
    }
  }
}

// ---------------------------------------------------------------------------
// Recurrence: 16 persistent blocks, each owns 64 columns of H (and of g).
// Weight slices (128 KB each) DMA'd once into LDS via the Tensor Data Mover.
// Cross-block sync: atomic counter (+ cluster barrier, NOP if not clustered).
// ---------------------------------------------------------------------------
__device__ inline void grid_sync(int* cnt, int target) {
  __syncthreads();
  if (threadIdx.x == 0) {
    __threadfence();
    atomicAdd(cnt, 1);
    while (__hip_atomic_load(cnt, __ATOMIC_RELAXED, __HIP_MEMORY_SCOPE_AGENT) <
           target) {
      __builtin_amdgcn_s_sleep(1);
    }
    __threadfence();
  }
  // Cluster user barrier: S_NOP when dispatch is not clustered (ISA 6.6);
  // called uniformly by every block so it is safe either way.
  __builtin_amdgcn_s_cluster_barrier();
  __syncthreads();
}

__global__ void k_recurrence(const __bf16* __restrict__ aT,   // [4R][H]
                             const __bf16* __restrict__ ctT,  // [H][4R]
                             const float* __restrict__ xb,    // [S*B][4R]
                             __bf16* __restrict__ h_glob,     // [B][H]
                             __bf16* __restrict__ g_glob,     // [B][4R]
                             __bf16* __restrict__ hseq,       // [S*B][H]
                             int* __restrict__ cnt,
                             float* __restrict__ out_c,       // [B][H]
                             float* __restrict__ out_h) {     // [B][H]
  extern __shared__ char smem[];
  __bf16* a_sl = (__bf16*)smem;                  // 64 x 1024 (128 KB)
  __bf16* ct_sl = (__bf16*)(smem + 64 * H_DIM * sizeof(__bf16));

  const int c0 = blockIdx.x * 64;  // owned column slice
  const int tid = threadIdx.x;     // 256
  const int wave = tid >> 5;       // 8 waves
  const int m_t = wave & 1;        // batch-tile (B=32 -> 2 tiles)
  const int n_t = wave >> 1;       // 0..3 -> 16-col subtile of slice
  const int lane = tid & 31;
  const int col = lane & 15;
  const int rbase = (lane >> 4) * 8;

  // TDM: DMA both weight slices into LDS (resident for all 128 steps).
  if (wave == 0) {
    tdm_load_64x1024_bf16(lds_addr32(a_sl),
                          (unsigned long long)(uintptr_t)(aT + (size_t)c0 * H_DIM));
    tdm_load_64x1024_bf16(lds_addr32(ct_sl),
                          (unsigned long long)(uintptr_t)(ctT + (size_t)c0 * R4));
    __builtin_amdgcn_s_wait_tensorcnt(0);
  }
  __syncthreads();

  float c_reg[8];
#pragma unroll
  for (int r = 0; r < 8; ++r) c_reg[r] = 0.0f;

  int phase = 0;
  for (int s = 0; s < S_LEN; ++s) {
    // ---- phase A: this block's 64-col slice of g = (h @ a) * xb_t ----
    v8f acc = vzero8();
    for (int kb = 0; kb < H_DIM / 32; ++kb) {
      v16bf af = frag_a(h_glob + (size_t)(m_t * 16) * H_DIM + kb * 32, H_DIM);
      v16bf bfr = frag_b(a_sl + (size_t)(n_t * 16) * H_DIM + kb * 32, H_DIM);
      acc = wmma_bf16(af, bfr, acc);
    }
    {
      const int gcol = c0 + n_t * 16 + col;
#pragma unroll
      for (int r = 0; r < 8; ++r) {
        const int bg = m_t * 16 + rbase + r;
        const float v = acc[r] * xb[((size_t)s * B_SZ + bg) * R4 + gcol];
        g_glob[(size_t)bg * R4 + gcol] = (__bf16)v;
      }
    }
    grid_sync(cnt, (++phase) * 16);

    // ---- phase B: 4 gate GEMMs over full g, LSTM pointwise update ----
    v8f ga[4];
#pragma unroll
    for (int k = 0; k < 4; ++k) {
      ga[k] = vzero8();
#pragma unroll
      for (int kb = 0; kb < RNK / 32; ++kb) {
        v16bf af = frag_a(
            g_glob + (size_t)(m_t * 16) * R4 + k * RNK + kb * 32, R4);
        v16bf bfr = frag_b(
            ct_sl + (size_t)(n_t * 16) * R4 + k * RNK + kb * 32, R4);
        ga[k] = wmma_bf16(af, bfr, ga[k]);
      }
    }
    {
      const int hcol = c0 + n_t * 16 + col;
#pragma unroll
      for (int r = 0; r < 8; ++r) {
        const int bg = m_t * 16 + rbase + r;
        const float ft = sigf(ga[0][r]);
        const float it = sigf(ga[1][r]);
        const float gt = tanhf(ga[2][r]);
        const float ot = sigf(ga[3][r]);
        c_reg[r] = ft * c_reg[r] + it * gt;
        const float hv = ot * tanhf(c_reg[r]);
        h_glob[(size_t)bg * H_DIM + hcol] = (__bf16)hv;
        hseq[((size_t)s * B_SZ + bg) * H_DIM + hcol] = (__bf16)hv;
        if (s == S_LEN - 1) {
          out_c[(size_t)bg * H_DIM + hcol] = c_reg[r];
          out_h[(size_t)bg * H_DIM + hcol] = hv;
        }
      }
    }
    grid_sync(cnt, (++phase) * 16);
  }
}

// ---------------------------------------------------------------------------
// Decoder: out[sb][v] = hseq[sb] . dec_w[v] + dec_b[v]
// (4096 x 32000, K=1024). Block tile 64M x 128N, 256 threads, 8 waves,
// wave = 16M x 64N. A tile async-staged; dec_w converted f32->bf16 per tile.
// ---------------------------------------------------------------------------
__global__ void k_decoder(const __bf16* __restrict__ hseq,
                          const float* __restrict__ dec_w,  // [V][H]
                          const float* __restrict__ dec_b,
                          float* __restrict__ out) {
  __shared__ __align__(16) char smem[(64 * 32 + 128 * 32) * sizeof(__bf16)];
  __bf16* as = (__bf16*)smem;                 // A tile  (ld 32)
  __bf16* bs = (__bf16*)(smem + 64 * 32 * 2); // B tile [n][k] (ld 32)
  const int tid = threadIdx.x;
  const int wave = tid >> 5;
  const int m_t = wave & 3;    // 0..3
  const int ngrp = wave >> 2;  // 0..1
  const int m0 = blockIdx.x * 64;
  const int n0 = blockIdx.y * 128;
  const int arow = tid >> 2, aseg = tid & 3;
  const uint32_t as_dst = lds_addr32(as + arow * 32 + aseg * 8);

  v8f acc[4];
#pragma unroll
  for (int t = 0; t < 4; ++t) acc[t] = vzero8();

  for (int kb = 0; kb < H_DIM / 32; ++kb) {
    const int k0 = kb * 32;
    // kick async A-tile fill
    async_copy_b128(as_dst,
                    hseq + (size_t)(m0 + arow) * H_DIM + k0 + aseg * 8);
    // stage B while the async fill is in flight (f32 -> bf16 into [n][k])
    {
      const int n = tid >> 1, half = tid & 1;
      const float* src = dec_w + (size_t)(n0 + n) * H_DIM + k0 + half * 16;
      if (kb + 1 < H_DIM / 32) __builtin_prefetch(src + 32, 0, 0);
      __bf16* dst = bs + n * 32 + half * 16;
#pragma unroll
      for (int i = 0; i < 16; ++i) dst[i] = (__bf16)src[i];
    }
    wait_async0();
    __syncthreads();
    v16bf af = frag_a(as + m_t * 16 * 32, 32);
#pragma unroll
    for (int nt = 0; nt < 4; ++nt) {
      v16bf bfr = frag_b(bs + ((ngrp * 4 + nt) * 16) * 32, 32);
      acc[nt] = wmma_bf16(af, bfr, acc[nt]);
    }
    __syncthreads();
  }

  const int lane = tid & 31;
  const int col = lane & 15;
  const int rbase = (lane >> 4) * 8;
#pragma unroll
  for (int nt = 0; nt < 4; ++nt) {
    const int nglob = n0 + (ngrp * 4 + nt) * 16 + col;
    const float bias = dec_b[nglob];
#pragma unroll
    for (int r = 0; r < 8; ++r) {
      const int rg = m0 + m_t * 16 + rbase + r;
      out[(size_t)rg * V_SZ + nglob] = acc[nt][r] + bias;
    }
  }
}

// ---------------------------------------------------------------------------
// Host-side launch
// ---------------------------------------------------------------------------
extern "C" void kernel_launch(void* const* d_in, const int* in_sizes, int n_in,
                              void* d_out, int out_size, void* d_ws,
                              size_t ws_size, hipStream_t stream) {
  (void)in_sizes; (void)n_in; (void)out_size; (void)ws_size;

  const int*   inp   = (const int*)d_in[0];
  const float* emb   = (const float*)d_in[1];
  const float* a     = (const float*)d_in[2];
  const float* b     = (const float*)d_in[3];
  const float* ct    = (const float*)d_in[4];
  const float* dec_w = (const float*)d_in[5];
  const float* dec_b = (const float*)d_in[6];

  char* ws = (char*)d_ws;
  size_t off = 0;
  __bf16* aT   = (__bf16*)(ws + off); off += (size_t)R4 * H_DIM * 2;           // 2 MB
  __bf16* bT   = (__bf16*)(ws + off); off += (size_t)R4 * D_DIM * 2;           // 2 MB
  __bf16* ctT  = (__bf16*)(ws + off); off += (size_t)H_DIM * R4 * 2;           // 2 MB
  __bf16* xbf  = (__bf16*)(ws + off); off += (size_t)S_LEN * B_SZ * D_DIM * 2; // 8 MB
  float*  xb   = (float*) (ws + off); off += (size_t)S_LEN * B_SZ * R4 * 4;    // 16 MB
  __bf16* hseq = (__bf16*)(ws + off); off += (size_t)S_LEN * B_SZ * H_DIM * 2; // 8 MB
  __bf16* h_g  = (__bf16*)(ws + off); off += (size_t)B_SZ * H_DIM * 2;         // 64 KB
  __bf16* g_g  = (__bf16*)(ws + off); off += (size_t)B_SZ * R4 * 2;            // 64 KB
  int*    cnt  = (int*)   (ws + off); off += 256;

  float* out   = (float*)d_out;
  float* out_c = out + (size_t)S_LEN * B_SZ * V_SZ;
  float* out_h = out_c + (size_t)B_SZ * H_DIM;

  dim3 tb(32, 8);
  k_transpose_cvt<<<dim3(R4 / 32, H_DIM / 32), tb, 0, stream>>>(a, aT, H_DIM, R4);
  k_transpose_cvt<<<dim3(R4 / 32, D_DIM / 32), tb, 0, stream>>>(b, bT, D_DIM, R4);
  k_transpose_cvt<<<dim3(H_DIM / 32, R4 / 32), tb, 0, stream>>>(ct, ctT, R4, H_DIM);

  k_gather<<<S_LEN * B_SZ, 256, 0, stream>>>(inp, emb, xbf);
  k_init<<<(B_SZ * H_DIM + 255) / 256, 256, 0, stream>>>(h_g, g_g, cnt);

  k_xb_gemm<<<dim3(S_LEN * B_SZ / 64, R4 / 64), 256, 0, stream>>>(xbf, bT, xb);

  k_recurrence<<<16, 256, 2 * 64 * 1024 * sizeof(__bf16), stream>>>(
      aT, ctT, xb, h_g, g_g, hseq, cnt, out_c, out_h);

  k_decoder<<<dim3(S_LEN * B_SZ / 64, V_SZ / 128), 256, 0, stream>>>(
      hseq, dec_w, dec_b, out);
}
